// ParserLSTM_19473381720573
// MI455X (gfx1250) — matrix operand order repaired and checked by reading, not compiled, not run on hardware
//
#include <hip/hip_runtime.h>

// ---------------------------------------------------------------------------
// Problem constants (match reference)
// ---------------------------------------------------------------------------
constexpr int T   = 512;
constexpr int Bsz = 64;
constexpr int I0  = 512;
constexpr int H   = 512;
constexpr int G   = 4 * H;   // 2048 gate columns

#define DEVFN __device__ __forceinline__

typedef __attribute__((ext_vector_type(16))) __bf16 v16bf;
typedef __attribute__((ext_vector_type(8)))  __bf16 v8bf;
typedef __attribute__((ext_vector_type(8)))  float  v8f;

// ---------------------------------------------------------------------------
// Helpers
// ---------------------------------------------------------------------------
DEVFN unsigned short f2bf(float f) {
  union { float f; unsigned u; } v; v.f = f;
  unsigned r = v.u + 0x7FFFu + ((v.u >> 16) & 1u);   // round-to-nearest-even
  return (unsigned short)(r >> 16);
}

DEVFN float fast_sigmoid(float x) {
  // 1/(1+2^(-x*log2 e)) via v_exp_f32 + v_rcp_f32 (TRANS32, co-executes w/ WMMA)
  return __builtin_amdgcn_rcpf(1.0f + __builtin_amdgcn_exp2f(-1.44269504f * x));
}
DEVFN float fast_tanh(float x) {
  return 2.0f * __builtin_amdgcn_rcpf(1.0f + __builtin_amdgcn_exp2f(-2.88539008f * x)) - 1.0f;
}

DEVFN v8f wmma_bf16(v16bf a, v16bf b, v8f c) {
  // D = A(16x32) * B(32x16) + C, f32 accumulate
  return __builtin_amdgcn_wmma_f32_16x16x32_bf16(false, a, false, b, (short)0, c, false, false);
}

// A-fragment: lane holds row M=lane&15; elements 0..15 = K[kb + h*16 .. +15],
// h = lane>>4. Two 16B LDS chunks.
DEVFN v16bf load_a16(const unsigned short* p) {
  v8bf lo = *(const v8bf*)(p);
  v8bf hi = *(const v8bf*)(p + 8);
  v16bf r;
#pragma unroll
  for (int i = 0; i < 8; i++) { r[i] = lo[i]; r[8 + i] = hi[i]; }
  return r;
}

// CDNA5 async global->LDS DMA (ASYNCcnt-tracked). lds_off is the wave-relative
// LDS byte offset (= low 32 bits of a generic pointer into __shared__).
DEVFN void async_ld128(unsigned lds_off, const void* gptr) {
  asm volatile("global_load_async_to_lds_b128 %0, %1, off"
               :: "v"(lds_off), "v"((unsigned long long)(size_t)gptr)
               : "memory");
}
DEVFN void wait_async0() { asm volatile("s_wait_asynccnt 0x0" ::: "memory"); }

// ---------------------------------------------------------------------------
// One K-region MAC sweep, software-pipelined:
//  - A fragments (LDS)  : 1 iteration ahead
//  - B fragments (L2)   : 2 iterations ahead
// Benign 64B over-read past the region end (never consumed; LDS OOB reads 0).
// abase: LDS activation base already adjusted for lane (row nlo, chunk hh*16)
// Brow : weight row pointer already adjusted for lane (col nlo, chunk hh*16)
// ---------------------------------------------------------------------------
template<int PITCH, int NK>
DEVFN void region_mac(const unsigned short* __restrict__ abase,
                      const unsigned short* __restrict__ Brow,
                      v8f acc[4]) {
  v16bf a[4], an[4];
  v16bf b0 = *(const v16bf*)(Brow + 0);
  v16bf b1 = *(const v16bf*)(Brow + 32);
#pragma unroll
  for (int m = 0; m < 4; m++) a[m] = load_a16(abase + m * 16 * PITCH);

#pragma unroll 4
  for (int kb = 0; kb < NK; kb += 32) {
    // prefetch next iteration's fragments into fresh registers
#pragma unroll
    for (int m = 0; m < 4; m++) an[m] = load_a16(abase + m * 16 * PITCH + kb + 32);
    v16bf bn = *(const v16bf*)(Brow + kb + 64);
    // consume current fragments (waits only on loads issued one iter ago)
#pragma unroll
    for (int m = 0; m < 4; m++) acc[m] = wmma_bf16(a[m], b0, acc[m]);
    b0 = b1; b1 = bn;
#pragma unroll
    for (int m = 0; m < 4; m++) a[m] = an[m];
  }
}

// Accumulate one gate (16x16 tiles for m_tile 0..3) over K = [x_t | h_prev]
template<int CIN>
DEVFN void gate_accum(const unsigned short* __restrict__ xb,     // LDS, 64 x CIN bf16
                      const unsigned short* __restrict__ hread,  // LDS, 64 x H bf16
                      const unsigned short* __restrict__ W,      // (2048 x KC) bf16 row-major
                      int gate, int wv, int lane,
                      float bias_val, v8f acc[4]) {
  constexpr int KC = CIN + H;
#pragma unroll
  for (int m = 0; m < 4; m++) {
#pragma unroll
    for (int e = 0; e < 8; e++) acc[m][e] = bias_val;   // bias depends on N only
  }
  const int nlo = lane & 15;
  const int hh  = lane >> 4;
  const unsigned short* Brow = W + (size_t)(gate * H + wv * 16 + nlo) * KC + hh * 16;
  region_mac<CIN, CIN>(xb    + nlo * CIN + hh * 16, Brow,       acc);
  region_mac<H,   H  >(hread + nlo * H   + hh * 16, Brow + CIN, acc);
}

// ---------------------------------------------------------------------------
// Prep kernels: f32 -> bf16 conversion / weight packing / bias folding
// ---------------------------------------------------------------------------
__global__ void cvt_bf16_kernel(const float* __restrict__ src,
                                unsigned short* __restrict__ dst, long long n) {
  for (long long i = (long long)blockIdx.x * blockDim.x + threadIdx.x; i < n;
       i += (long long)gridDim.x * blockDim.x)
    dst[i] = f2bf(src[i]);
}

// Pack W_cat = [W_ih | W_hh] per gate-row: (2048 x (Cin+512)) bf16
__global__ void pack_w_kernel(const float* __restrict__ wih,
                              const float* __restrict__ whh,
                              unsigned short* __restrict__ dst, int Cin) {
  const int Kc = Cin + H;
  const long long total = (long long)G * Kc;
  for (long long i = (long long)blockIdx.x * blockDim.x + threadIdx.x; i < total;
       i += (long long)gridDim.x * blockDim.x) {
    int g = (int)(i / Kc), c = (int)(i % Kc);
    float v = (c < Cin) ? wih[(size_t)g * Cin + c] : whh[(size_t)g * H + (c - Cin)];
    dst[i] = f2bf(v);
  }
}

__global__ void bias_sum_kernel(const float* __restrict__ bih,
                                const float* __restrict__ bhh,
                                float* __restrict__ dst) {
  int i = blockIdx.x * blockDim.x + threadIdx.x;
  if (i < G) dst[i] = bih[i] + bhh[i];
}

// ---------------------------------------------------------------------------
// Recurrent scan: one block (32 waves) per direction; grid.x = 2 (fwd, bwd)
// LDS: double-buffered h (2*64KB) + staged x_t (64/128KB) + mask row
// ---------------------------------------------------------------------------
template<int CIN>
__global__ __launch_bounds__(1024)
void lstm_scan_kernel(const unsigned short* __restrict__ act,            // (T,64,CIN) bf16
                      const unsigned short* __restrict__ Wf,
                      const unsigned short* __restrict__ Wb,
                      const float* __restrict__ biasF,
                      const float* __restrict__ biasB,
                      const float* __restrict__ mask,                    // (T,64) f32
                      unsigned short* __restrict__ out_bf, int pitch_bf, // optional bf16 out
                      float* __restrict__ out_f32, int pitch_f)          // optional f32 out
{
  const int dir  = blockIdx.x;
  const bool rev = (dir == 1);
  const unsigned short* W = rev ? Wb : Wf;
  const float* bias = rev ? biasB : biasF;
  const int coloff = dir * H;

  extern __shared__ unsigned char smem[];
  unsigned short* hb = (unsigned short*)smem;          // 2 * 64*512 bf16
  unsigned short* xb = hb + 2 * Bsz * H;               // 64 * CIN bf16
  float* mb = (float*)(xb + (size_t)Bsz * CIN);        // 64 f32
  const unsigned xb_off = (unsigned)(size_t)(void*)xb; // wave-relative LDS byte offset

  const int tid  = threadIdx.x;
  const int lane = tid & 31;
  const int wv   = tid >> 5;        // wave id == N-tile id (0..31)
  const int nlo  = lane & 15;
  const int hh   = lane >> 4;
  const int ncol = wv * 16 + nlo;

  // h0 = 0
  for (int i = tid; i < Bsz * H; i += 1024) hb[i] = 0;

  float bias_g[4];
#pragma unroll
  for (int g = 0; g < 4; g++) bias_g[g] = bias[g * H + ncol];

  v8f cst[4];
#pragma unroll
  for (int m = 0; m < 4; m++)
#pragma unroll
    for (int e = 0; e < 8; e++) cst[m][e] = 0.0f;

  for (int step = 0; step < T; step++) {
    const int t = rev ? (T - 1 - step) : step;
    __syncthreads();   // previous step done reading xb / writing h

    // stage x_t into LDS via async global->LDS DMA (16B per op per lane)
    {
      const unsigned short* src = act + (size_t)t * Bsz * CIN;
      constexpr int n16 = Bsz * CIN / 8;      // # of 16B chunks
      for (int i = tid; i < n16; i += 1024)
        async_ld128(xb_off + (unsigned)i * 16, src + (size_t)i * 8);
      if (tid < Bsz) mb[tid] = mask[(size_t)t * Bsz + tid];
      // prefetch next timestep's activation row toward L2
      if (step + 1 < T) {
        const int tn = rev ? (T - 2 - step) : (step + 1);
        const unsigned short* nsrc = act + (size_t)tn * Bsz * CIN;
        if (tid * 64 < Bsz * CIN) __builtin_prefetch(nsrc + tid * 64, 0, 0);
      }
      wait_async0();   // our async writes to LDS landed
    }
    __syncthreads();   // everyone's staging visible

    const unsigned short* hread  = hb + (size_t)(step & 1) * Bsz * H;
    unsigned short*       hwrite = hb + (size_t)((step + 1) & 1) * Bsz * H;

    v8f ga[4], gb[4], ig[4];
    // i-gate and g-gate -> ig = sigmoid(i)*tanh(g)
    gate_accum<CIN>(xb, hread, W, 0, wv, lane, bias_g[0], ga);
    gate_accum<CIN>(xb, hread, W, 2, wv, lane, bias_g[2], gb);
#pragma unroll
    for (int m = 0; m < 4; m++)
#pragma unroll
      for (int e = 0; e < 8; e++)
        ig[m][e] = fast_sigmoid(ga[m][e]) * fast_tanh(gb[m][e]);

    // f-gate -> c = sigmoid(f)*c + ig
    gate_accum<CIN>(xb, hread, W, 1, wv, lane, bias_g[1], ga);
#pragma unroll
    for (int m = 0; m < 4; m++)
#pragma unroll
      for (int e = 0; e < 8; e++)
        cst[m][e] = fast_sigmoid(ga[m][e]) * cst[m][e] + ig[m][e];

    // o-gate -> h = sigmoid(o)*tanh(c); apply mask; emit h
    gate_accum<CIN>(xb, hread, W, 3, wv, lane, bias_g[3], gb);
#pragma unroll
    for (int m = 0; m < 4; m++) {
#pragma unroll
      for (int e = 0; e < 8; e++) {
        const int b = m * 16 + hh * 8 + e;          // C/D layout: M = r + 8*(lane>>4)
        const float mv = mb[b];
        const float hn = fast_sigmoid(gb[m][e]) * fast_tanh(cst[m][e]) * mv;
        cst[m][e] *= mv;
        hwrite[(size_t)b * H + ncol] = f2bf(hn);
        const size_t orow = (size_t)t * Bsz + b;
        if (out_bf)  out_bf[orow * pitch_bf + coloff + ncol] = f2bf(hn);
        if (out_f32) out_f32[orow * pitch_f + coloff + ncol] = hn;
      }
    }
  }
}

// ---------------------------------------------------------------------------
// Host launcher
// ---------------------------------------------------------------------------
extern "C" void kernel_launch(void* const* d_in, const int* in_sizes, int n_in,
                              void* d_out, int out_size, void* d_ws, size_t ws_size,
                              hipStream_t stream) {
  const float* x      = (const float*)d_in[0];
  const float* mask   = (const float*)d_in[1];
  const float* f_wih0 = (const float*)d_in[2];
  const float* f_whh0 = (const float*)d_in[3];
  const float* f_bih0 = (const float*)d_in[4];
  const float* f_bhh0 = (const float*)d_in[5];
  const float* b_wih0 = (const float*)d_in[6];
  const float* b_whh0 = (const float*)d_in[7];
  const float* b_bih0 = (const float*)d_in[8];
  const float* b_bhh0 = (const float*)d_in[9];
  const float* f_wih1 = (const float*)d_in[10];
  const float* f_whh1 = (const float*)d_in[11];
  const float* f_bih1 = (const float*)d_in[12];
  const float* f_bhh1 = (const float*)d_in[13];
  const float* b_wih1 = (const float*)d_in[14];
  const float* b_whh1 = (const float*)d_in[15];
  const float* b_bih1 = (const float*)d_in[16];
  const float* b_bhh1 = (const float*)d_in[17];
  (void)in_sizes; (void)n_in; (void)out_size; (void)ws_size;

  unsigned char* ws = (unsigned char*)d_ws;
  size_t off = 0;
  auto carve = [&](size_t bytes) -> void* {
    void* p = ws + off;
    off += (bytes + 255) & ~(size_t)255;
    return p;
  };

  unsigned short* xin  = (unsigned short*)carve((size_t)T * Bsz * I0 * 2);      // 32 MB
  unsigned short* w0f  = (unsigned short*)carve((size_t)G * (I0 + H) * 2);      // 4 MB
  unsigned short* w0b  = (unsigned short*)carve((size_t)G * (I0 + H) * 2);      // 4 MB
  unsigned short* w1f  = (unsigned short*)carve((size_t)G * (2 * H + H) * 2);   // 6 MB
  unsigned short* w1b  = (unsigned short*)carve((size_t)G * (2 * H + H) * 2);   // 6 MB
  float*          bsum = (float*)carve((size_t)4 * G * sizeof(float));          // 32 KB
  unsigned short* l0o  = (unsigned short*)carve((size_t)T * Bsz * 2 * H * 2);   // 64 MB

  float* b0f = bsum + 0 * G;
  float* b0b = bsum + 1 * G;
  float* b1f = bsum + 2 * G;
  float* b1b = bsum + 3 * G;

  // 1) x -> bf16
  cvt_bf16_kernel<<<8192, 256, 0, stream>>>(x, xin, (long long)T * Bsz * I0);

  // 2) pack [W_ih | W_hh] -> bf16, fold biases
  pack_w_kernel<<<2048, 256, 0, stream>>>(f_wih0, f_whh0, w0f, I0);
  pack_w_kernel<<<2048, 256, 0, stream>>>(b_wih0, b_whh0, w0b, I0);
  pack_w_kernel<<<3072, 256, 0, stream>>>(f_wih1, f_whh1, w1f, 2 * H);
  pack_w_kernel<<<3072, 256, 0, stream>>>(b_wih1, b_whh1, w1b, 2 * H);
  bias_sum_kernel<<<8, 256, 0, stream>>>(f_bih0, f_bhh0, b0f);
  bias_sum_kernel<<<8, 256, 0, stream>>>(b_bih0, b_bhh0, b0b);
  bias_sum_kernel<<<8, 256, 0, stream>>>(f_bih1, f_bhh1, b1f);
  bias_sum_kernel<<<8, 256, 0, stream>>>(b_bih1, b_bhh1, b1b);

  // 3) layer 0 scan (both directions concurrently), bf16 output (layer-1 input)
  {
    size_t sh = (size_t)2 * Bsz * H * 2 + (size_t)Bsz * I0 * 2 + Bsz * sizeof(float);
    lstm_scan_kernel<I0><<<2, 1024, sh, stream>>>(xin, w0f, w0b, b0f, b0b, mask,
                                                  l0o, 2 * H, nullptr, 0);
  }
  // 4) layer 1 scan, f32 output straight to d_out (T,64,1024)
  {
    size_t sh = (size_t)2 * Bsz * H * 2 + (size_t)Bsz * (2 * H) * 2 + Bsz * sizeof(float);
    lstm_scan_kernel<2 * H><<<2, 1024, sh, stream>>>(l0o, w1f, w1b, b1f, b1b, mask,
                                                     nullptr, 0, (float*)d_out, 2 * H);
  }
}
